// VQModel_49074296324149
// MI455X (gfx1250) — matrix operand order repaired and checked by reading, not compiled, not run on hardware
//
#include <hip/hip_runtime.h>
#include <hip/hip_bf16.h>

// ---------------------------------------------------------------------------
// VQ model forward on MI455X (gfx1250, wave32, WMMA).
//
// Roofline: dominant op is logits = relu(x*w1+b1) @ W2^T:
//   2 * 262144 * 512 * 128 = 34.4 GFLOP vs ~3.4 MB of real HBM traffic
// -> pure matrix-core problem. softmax is dead (STE collapses to hard
// one-hot in forward value), decoder collapses to a 512-entry scalar LUT.
//
// v5 changes: v4's scheduler sank the up-front fragment loads back to a
// single just-in-time B buffer (s_wait_loadcnt 0x0 before every WMMA pair).
// Now a sched_barrier(0) sits between each N-tile's loads and its WMMA
// chain; with the nt loop unrolled by 2 this creates regions where tile
// nt+1's loads can only live ABOVE tile nt's WMMA chain -> each tile's
// 8 loads are issued a full 8-WMMA chain before first use.
// ---------------------------------------------------------------------------

typedef __attribute__((ext_vector_type(16))) __bf16 v16bf;
typedef __attribute__((ext_vector_type(8)))  float  v8f;

#define HIDDEN        128
#define NUM_CODES     512
#define NTILES        (NUM_CODES / 16)   // 32
#define KTILES        (HIDDEN / 32)      // 4
#define WAVES         8
#define ROWS_PER_WAVE 32                 // two 16-row M tiles per wave
#define ROWS_PER_ITER (WAVES * ROWS_PER_WAVE)   // 256
#define ITERS         2
#define ROWS_PER_BLOCK (ROWS_PER_ITER * ITERS)  // 512

static __device__ __forceinline__ unsigned short f32_to_bf16_rne(float f) {
    unsigned int u = __float_as_uint(f);
    u += 0x7FFFu + ((u >> 16) & 1u);   // round-to-nearest-even
    return (unsigned short)(u >> 16);
}

// --- prep1: pack W2 into WMMA-B fragment-major bf16 -------------------------
// Fragment f = nt*KTILES + kt, 1 KB each: lane l owns 32 contiguous bytes.
// B layout (16-bit, 32x16): lanes 0-15 hold K=0..7,16..23, lanes 16-31 hold
// K=8..15,24..31; halfword i of the lane's 8 VGPRs is K = base + (i<8 ? i : i+8). Column
// N = nt*16 + (l & 15).
__global__ void pack_w2_kernel(const float* __restrict__ w2,
                               unsigned short* __restrict__ packed) {
    int f  = blockIdx.x;            // 0..127
    int nt = f >> 2, kt = f & 3;
    int l  = threadIdx.x;           // 0..31
    int n  = nt * 16 + (l & 15);
    int kb = kt * 32 + ((l >= 16) ? 8 : 0);
    const float* src = w2 + n * HIDDEN + kb;     // w2 is [code][hidden], k-contig
    unsigned short* dst = packed + f * 512 + l * 16;
#pragma unroll
    for (int i = 0; i < 16; ++i) dst[i] = f32_to_bf16_rne(src[i < 8 ? i : i + 8]);
}

// --- prep2: score LUT -------------------------------------------------------
__global__ void score_kernel(const float* __restrict__ emb,
                             const float* __restrict__ dec_w,
                             const float* __restrict__ dec_b,
                             float* __restrict__ score) {
    int c = blockIdx.x * blockDim.x + threadIdx.x;
    if (c < NUM_CODES) {
        const float* e = emb + c * 256;
        float s = 0.f;
#pragma unroll 8
        for (int d = 0; d < 256; ++d) s = fmaf(e[d], dec_w[d], s);
        score[c] = s + dec_b[0];
    }
}

// --- main fused kernel ------------------------------------------------------
__global__ __launch_bounds__(WAVES * 32)
__attribute__((amdgpu_waves_per_eu(4)))
void vq_fused_kernel(const float* __restrict__ x,
                     const float* __restrict__ w1,
                     const float* __restrict__ b1,
                     const float* __restrict__ b2,
                     const v16bf* __restrict__ w2frag,
                     const float* __restrict__ score,
                     float* __restrict__ out,
                     int*  __restrict__ idxOut) {
    __shared__ float w1s[HIDDEN], b1s[HIDDEN], b2s[NUM_CODES], scoreS[NUM_CODES];
    int t = threadIdx.x;
    if (t < HIDDEN) { w1s[t] = w1[t]; b1s[t] = b1[t]; }
    for (int i = t; i < NUM_CODES; i += WAVES * 32) { b2s[i] = b2[i]; scoreS[i] = score[i]; }
    __syncthreads();

    const int lane = t & 31;
    const int wave = t >> 5;
    const int col  = lane & 15;
    const int hi16 = lane >> 4;     // A-layout: lanes 16-31 own K offset +8

#pragma unroll 1
    for (int it = 0; it < ITERS; ++it) {
        const int rowBase = blockIdx.x * ROWS_PER_BLOCK + it * ROWS_PER_ITER
                          + wave * ROWS_PER_WAVE;
        const float xv0 = x[rowBase + col];        // M tile 0: rows rowBase..+15
        const float xv1 = x[rowBase + 16 + col];   // M tile 1: rows rowBase+16..+31

        // Synthesize A fragments in registers (16-bit A layout, 16x32 per chunk):
        // lane<16: halfword i -> K = kb + (i<8 ? i : i+8); lane>=16: +8.
        // w1/b1 pairs are (k0, k0+1) with k0 even -> one ds_load_b64 each.
        union AU { v16bf v; unsigned int u[8]; };
        AU a0[KTILES], a1[KTILES];
        const float2* w1p = (const float2*)w1s;
        const float2* b1p = (const float2*)b1s;
#pragma unroll
        for (int kt = 0; kt < KTILES; ++kt) {
            const int kb = kt * 32 + hi16 * 8;
#pragma unroll
            for (int v = 0; v < 8; ++v) {
                const int i0 = 2 * v;
                const int k0 = kb + (i0 < 8 ? i0 : i0 + 8);   // always even
                const float2 w = w1p[k0 >> 1];
                const float2 bb = b1p[k0 >> 1];
                float h00 = fmaxf(fmaf(xv0, w.x, bb.x), 0.f);
                float h01 = fmaxf(fmaf(xv0, w.y, bb.y), 0.f);
                float h10 = fmaxf(fmaf(xv1, w.x, bb.x), 0.f);
                float h11 = fmaxf(fmaf(xv1, w.y, bb.y), 0.f);
                a0[kt].u[v] = (unsigned)f32_to_bf16_rne(h00) | ((unsigned)f32_to_bf16_rne(h01) << 16);
                a1[kt].u[v] = (unsigned)f32_to_bf16_rne(h10) | ((unsigned)f32_to_bf16_rne(h11) << 16);
            }
            // cap live ranges: don't let the scheduler hoist the next chunk's
            // LDS loads over this chunk's packing
            __builtin_amdgcn_sched_barrier(0);
        }

        // Running argmax over the 32 N-tiles, kept in the f32 C/D layout:
        // element e of lane -> row = rowBase(+16) + hi16*8 + e, column = col.
        float mv0[8], mv1[8];
        int   mi0[8], mi1[8];
#pragma unroll
        for (int e = 0; e < 8; ++e) {
            mv0[e] = -__builtin_inff(); mv1[e] = -__builtin_inff();
            mi0[e] = 0;                 mi1[e] = 0;
        }

        // unroll-by-2 + sched_barrier between loads and WMMAs: region shape
        //   [loads nt] | [loads nt+1 ... wmma nt x8, argmax nt] | [wmma nt+1 x8 ...]
        // so every tile's loads issue a full WMMA chain before first use.
#pragma unroll 2
        for (int nt = 0; nt < NTILES; ++nt) {
            // Fold the enc_b2 bias into the accumulator: D = A*B + C, C = bias.
            const float bias = b2s[nt * 16 + col];
            const v16bf* fp = w2frag + nt * (KTILES * 32) + lane;
            v16bf bf0 = fp[0 * 32];
            v16bf bf1 = fp[1 * 32];
            v16bf bf2 = fp[2 * 32];
            v16bf bf3 = fp[3 * 32];
            v8f c0, c1;
#pragma unroll
            for (int e = 0; e < 8; ++e) { c0[e] = bias; c1[e] = bias; }
            // pipeline pin: loads above stay above; WMMAs below can't pull them down
            __builtin_amdgcn_sched_barrier(0);
            c0 = __builtin_amdgcn_wmma_f32_16x16x32_bf16(false, a0[0].v, false, bf0, (short)0, c0, false, false);
            c1 = __builtin_amdgcn_wmma_f32_16x16x32_bf16(false, a1[0].v, false, bf0, (short)0, c1, false, false);
            c0 = __builtin_amdgcn_wmma_f32_16x16x32_bf16(false, a0[1].v, false, bf1, (short)0, c0, false, false);
            c1 = __builtin_amdgcn_wmma_f32_16x16x32_bf16(false, a1[1].v, false, bf1, (short)0, c1, false, false);
            c0 = __builtin_amdgcn_wmma_f32_16x16x32_bf16(false, a0[2].v, false, bf2, (short)0, c0, false, false);
            c1 = __builtin_amdgcn_wmma_f32_16x16x32_bf16(false, a1[2].v, false, bf2, (short)0, c1, false, false);
            c0 = __builtin_amdgcn_wmma_f32_16x16x32_bf16(false, a0[3].v, false, bf3, (short)0, c0, false, false);
            c1 = __builtin_amdgcn_wmma_f32_16x16x32_bf16(false, a1[3].v, false, bf3, (short)0, c1, false, false);
            const int n = nt * 16 + col;
#pragma unroll
            for (int e = 0; e < 8; ++e) {
                if (c0[e] > mv0[e]) { mv0[e] = c0[e]; mi0[e] = n; }  // strict > keeps first max
                if (c1[e] > mv1[e]) { mv1[e] = c1[e]; mi1[e] = n; }
            }
        }

        // Tree-reduce argmax across the 16 lanes of each half-wave
        // (xor masks 1,2,4,8 never cross the 16-lane boundary).
#pragma unroll
        for (int off = 8; off >= 1; off >>= 1) {
#pragma unroll
            for (int e = 0; e < 8; ++e) {
                float ov; int oi;
                ov = __shfl_xor(mv0[e], off, 32); oi = __shfl_xor(mi0[e], off, 32);
                if (ov > mv0[e] || (ov == mv0[e] && oi < mi0[e])) { mv0[e] = ov; mi0[e] = oi; }
                ov = __shfl_xor(mv1[e], off, 32); oi = __shfl_xor(mi1[e], off, 32);
                if (ov > mv1[e] || (ov == mv1[e] && oi < mi1[e])) { mv1[e] = ov; mi1[e] = oi; }
            }
        }

        if (col == 0) {   // lanes 0 (rows e) and 16 (rows 8+e) write results
            const int base0 = rowBase + hi16 * 8;
            const int base1 = rowBase + 16 + hi16 * 8;
#pragma unroll
            for (int e = 0; e < 8; ++e) {
                out[base0 + e]    = scoreS[mi0[e]];
                idxOut[base0 + e] = mi0[e];
                out[base1 + e]    = scoreS[mi1[e]];
                idxOut[base1 + e] = mi1[e];
            }
        }
    }
}

extern "C" void kernel_launch(void* const* d_in, const int* in_sizes, int n_in,
                              void* d_out, int out_size, void* d_ws, size_t ws_size,
                              hipStream_t stream) {
    const float* x     = (const float*)d_in[0];
    const float* w1    = (const float*)d_in[1];
    const float* b1    = (const float*)d_in[2];
    const float* w2    = (const float*)d_in[3];
    const float* b2    = (const float*)d_in[4];
    const float* emb   = (const float*)d_in[5];
    const float* dec_w = (const float*)d_in[6];
    const float* dec_b = (const float*)d_in[7];
    const int B = in_sizes[0];   // x is [B,1]

    // workspace: [0,128KB) packed W2 fragments (bf16), then 512-float score LUT
    unsigned short* packed = (unsigned short*)d_ws;
    float* score = (float*)((char*)d_ws + NUM_CODES * HIDDEN * sizeof(unsigned short));

    pack_w2_kernel<<<NTILES * KTILES, 32, 0, stream>>>(w2, packed);
    score_kernel<<<2, 256, 0, stream>>>(emb, dec_w, dec_b, score);

    float* outv = (float*)d_out;        // out[B] f32 ...
    int*   idxo = (int*)d_out + B;      // ... then hard_idx[B] i32

    vq_fused_kernel<<<B / ROWS_PER_BLOCK, WAVES * 32, 0, stream>>>(
        x, w1, b1, b2, (const v16bf*)packed, score, outv, idxo);
}
